// GraphNetBlock_84619445666102
// MI455X (gfx1250) — compile-verified, matrix operand order checked by reading, not run on hardware
//
#include <hip/hip_runtime.h>
#include <hip/hip_bf16.h>

#define NN 4096
#define DD 128
#define HH 4
#define HID 32
#define EPSF 1e-5f
#define SLOPE 0.2f

typedef __attribute__((ext_vector_type(2)))  float  v2f;
typedef __attribute__((ext_vector_type(8)))  float  v8f;
typedef __attribute__((ext_vector_type(16))) __bf16 v16bf;

__device__ __forceinline__ __bf16 f2bf(float f) {
  unsigned u = __builtin_bit_cast(unsigned, f);
  unsigned r = u + 0x7fffu + ((u >> 16) & 1u);         // round-to-nearest-even
  unsigned short s = (unsigned short)(r >> 16);
  return __builtin_bit_cast(__bf16, s);
}

__device__ __forceinline__ float lrelu(float x) { return x >= 0.f ? x : SLOPE * x; }

// K index of element j within a 16-element bf16 WMMA operand (16x32 A / 32x16 B),
// per CDNA5 ISA 16-bit operand layout (lane-half selects K group of 8).
__device__ __forceinline__ int kOf(int j, int lane) {
  return ((lane >> 4) << 3) + j + ((j >= 8) ? 8 : 0);
}

// ---------------------------------------------------------------------------
// Kernel 1: fused projections (fp32 WMMA 16x16x4), lrelu+LN+residual epilogue,
// plus bf16 staging of Q, S, S^T and the scaled q·r term for attention.
// ---------------------------------------------------------------------------
__global__ __launch_bounds__(256)
void proj_kernel(const float* __restrict__ node, const float* __restrict__ recv,
                 const float* __restrict__ send,
                 const float* __restrict__ Wq, const float* __restrict__ Wr,
                 const float* __restrict__ br, const float* __restrict__ Ws,
                 const float* __restrict__ bs, const float* __restrict__ Wm,
                 const float* __restrict__ bm,
                 const float* __restrict__ g_send, const float* __restrict__ b_send,
                 const float* __restrict__ g_recv, const float* __restrict__ b_recv,
                 float* __restrict__ out_recv, float* __restrict__ out_send,
                 __bf16* __restrict__ qbf, __bf16* __restrict__ sbf,
                 __bf16* __restrict__ stb, float* __restrict__ qrs)
{
  __shared__ float bufN[16 * DD];
  __shared__ float bufR[16 * DD];
  __shared__ float bufS[16 * DD];
  const int t  = threadIdx.x;
  const int n0 = blockIdx.x * 16;

  // stage 16x128 input tiles (coalesced)
  for (int i = 0; i < 8; ++i) {
    int idx = t * 8 + i;                      // 0..2047
    bufN[idx] = node[n0 * DD + idx];
    bufR[idx] = recv[n0 * DD + idx];
    bufS[idx] = send[n0 * DD + idx];
  }
  __syncthreads();

  const int lane = t & 31;
  const int w    = t >> 5;                    // 16-col tile per wave
  const int colL = lane & 15;
  const int col  = w * 16 + colL;
  const int rowA = lane & 15;
  const int hi   = lane >> 4;

  v8f c1 = {0,0,0,0,0,0,0,0};
  v8f c2 = {0,0,0,0,0,0,0,0};
  v8f c3 = {0,0,0,0,0,0,0,0};
  for (int ks = 0; ks < 32; ++ks) {
    int k0 = ks * 4 + hi * 2;
    v2f aN, aR, aS, bRr, bSs, bM, bQ;
    aN.x = bufN[rowA * DD + k0]; aN.y = bufN[rowA * DD + k0 + 1];
    aR.x = bufR[rowA * DD + k0]; aR.y = bufR[rowA * DD + k0 + 1];
    aS.x = bufS[rowA * DD + k0]; aS.y = bufS[rowA * DD + k0 + 1];
    bRr.x = Wr[k0 * DD + col]; bRr.y = Wr[(k0 + 1) * DD + col];
    bSs.x = Ws[k0 * DD + col]; bSs.y = Ws[(k0 + 1) * DD + col];
    bM.x  = Wm[k0 * DD + col]; bM.y  = Wm[(k0 + 1) * DD + col];
    bQ.x  = Wq[k0 * DD + col]; bQ.y  = Wq[(k0 + 1) * DD + col];
    c1 = __builtin_amdgcn_wmma_f32_16x16x4_f32(false, aN, false, bRr, (short)0, c1, false, false);
    c1 = __builtin_amdgcn_wmma_f32_16x16x4_f32(false, aR, false, bM,  (short)0, c1, false, false);
    c2 = __builtin_amdgcn_wmma_f32_16x16x4_f32(false, aN, false, bSs, (short)0, c2, false, false);
    c2 = __builtin_amdgcn_wmma_f32_16x16x4_f32(false, aS, false, bM,  (short)0, c2, false, false);
    c3 = __builtin_amdgcn_wmma_f32_16x16x4_f32(false, aN, false, bQ,  (short)0, c3, false, false);
  }
  float bias1 = br[col] + bm[col];
  float bias2 = bs[col] + bm[col];

  __syncthreads();
  // overwrite staging: bufR <- lrelu(P_recv), bufS <- lrelu(P_send), bufN <- Q
  for (int r = 0; r < 8; ++r) {
    int row = r + hi * 8;                     // C layout: VGPR r -> rows r / r+8
    bufR[row * DD + col] = lrelu(c1[r] + bias1);
    bufS[row * DD + col] = lrelu(c2[r] + bias2);
    bufN[row * DD + col] = c3[r];
  }
  __syncthreads();

  // LayerNorm epilogue: 16 threads per row, 8 cols each
  {
    int row = t >> 4;
    int li  = t & 15;
    int c0  = li * 8;
    float x1[8], x2[8], q[8], rv[8], sv[8];
    float s1 = 0.f, s2 = 0.f;
    for (int i = 0; i < 8; ++i) {
      x1[i] = bufR[row * DD + c0 + i];
      x2[i] = bufS[row * DD + c0 + i];
      q[i]  = bufN[row * DD + c0 + i];
      rv[i] = recv[(n0 + row) * DD + c0 + i];
      sv[i] = send[(n0 + row) * DD + c0 + i];
      s1 += x1[i]; s2 += x2[i];
    }
    for (int m = 1; m < 16; m <<= 1) { s1 += __shfl_xor(s1, m, 16); s2 += __shfl_xor(s2, m, 16); }
    float m1 = s1 / DD, m2 = s2 / DD;
    float v1 = 0.f, v2 = 0.f;
    for (int i = 0; i < 8; ++i) { float d1 = x1[i] - m1, d2 = x2[i] - m2; v1 += d1 * d1; v2 += d2 * d2; }
    for (int m = 1; m < 16; m <<= 1) { v1 += __shfl_xor(v1, m, 16); v2 += __shfl_xor(v2, m, 16); }
    float rs1 = rsqrtf(v1 / DD + EPSF), rs2 = rsqrtf(v2 / DD + EPSF);
    float qsum = 0.f;
    for (int i = 0; i < 8; ++i) {
      int c = c0 + i;
      out_recv[(n0 + row) * DD + c] = (x1[i] - m1) * rs1 * g_recv[c] + b_recv[c] + rv[i];
      out_send[(n0 + row) * DD + c] = (x2[i] - m2) * rs2 * g_send[c] + b_send[c] + sv[i];
      qbf[(n0 + row) * DD + c] = f2bf(q[i]);
      __bf16 sb = f2bf(sv[i]);
      sbf[(n0 + row) * DD + c] = sb;           // S bf16, [m][d] for PV
      stb[c * NN + (n0 + row)] = sb;           // S^T bf16, [H*HID][N] for QK^T
      qsum += q[i] * rv[i];
    }
    qsum += __shfl_xor(qsum, 1, 4);
    qsum += __shfl_xor(qsum, 2, 4);
    if ((li & 3) == 0) {
      int h = li >> 2;
      qrs[h * NN + n0 + row] = qsum * 0.1767766952966369f;  // pre-scaled by 1/sqrt(HID)
    }
  }
}

// ---------------------------------------------------------------------------
// Kernel 2: fused masked multi-head attention (flash-style, bf16 WMMA 16x16x32)
// + lrelu + LayerNorm + residual. One wave per head, 16 query rows per block.
// ---------------------------------------------------------------------------
__global__ __launch_bounds__(128)
void attn_kernel(const float* __restrict__ node, const float* __restrict__ recv,
                 const float* __restrict__ adj,
                 const __bf16* __restrict__ qbf, const __bf16* __restrict__ sbf,
                 const __bf16* __restrict__ stb, const float* __restrict__ qrs,
                 const float* __restrict__ g_node, const float* __restrict__ b_node,
                 float* __restrict__ out_node)
{
  __shared__ __bf16 pbuf[HH][16 * 32];
  __shared__ float  obuf[16 * DD];
  const int t     = threadIdx.x;
  const int lane  = t & 31;
  const int h     = t >> 5;                   // head == wave
  const int n0    = blockIdx.x * 16;
  const int colL  = lane & 15;
  const int hiOff = (lane >> 4) * 8;
  const float NEG_INF = -__builtin_inff();
  const float invsq   = 0.1767766952966369f;  // 1/sqrt(32)

  // Q A-operand (16x32 bf16), loop-invariant
  v16bf qa;
  {
    int row = n0 + (lane & 15);
    for (int j = 0; j < 16; ++j)
      qa[j] = qbf[row * DD + h * HID + kOf(j, lane)];
  }
  float qrr[8];
  for (int r = 0; r < 8; ++r) qrr[r] = qrs[h * NN + n0 + r + hiOff];

  float m_i[8], l_i[8];
  v8f acc0 = {0,0,0,0,0,0,0,0};
  v8f acc1 = {0,0,0,0,0,0,0,0};
  for (int r = 0; r < 8; ++r) { m_i[r] = NEG_INF; l_i[r] = 0.f; }

  for (int mb = 0; mb < NN; mb += 32) {
    // B operands from S^T: K=d (stride N), cols m contiguous
    v16bf b0, b1;
    for (int j = 0; j < 16; ++j) {
      int k = kOf(j, lane);
      b0[j] = stb[(h * HID + k) * NN + mb + colL];
      b1[j] = stb[(h * HID + k) * NN + mb + 16 + colL];
    }
    v8f z = {0,0,0,0,0,0,0,0};
    v8f s0 = __builtin_amdgcn_wmma_f32_16x16x32_bf16(false, qa, false, b0, (short)0, z, false, false);
    v8f s1 = __builtin_amdgcn_wmma_f32_16x16x32_bf16(false, qa, false, b1, (short)0, z, false, false);

    for (int r = 0; r < 8; ++r) {
      size_t row = (size_t)(n0 + r + hiOff);
      float a0 = adj[row * NN + mb + colL];
      float a1 = adj[row * NN + mb + 16 + colL];
      float sv0 = (a0 > 0.f) ? (s0[r] * invsq + qrr[r]) : NEG_INF;
      float sv1 = (a1 > 0.f) ? (s1[r] * invsq + qrr[r]) : NEG_INF;
      float mx = fmaxf(sv0, sv1);
      for (int m = 1; m < 16; m <<= 1) mx = fmaxf(mx, __shfl_xor(mx, m, 16));
      float mnew = fmaxf(m_i[r], mx);
      float al = (m_i[r] == mnew) ? 1.f : __expf(m_i[r] - mnew);
      float p0 = (mnew > NEG_INF) ? __expf(sv0 - mnew) : 0.f;
      float p1 = (mnew > NEG_INF) ? __expf(sv1 - mnew) : 0.f;
      float ps = p0 + p1;
      for (int m = 1; m < 16; m <<= 1) ps += __shfl_xor(ps, m, 16);
      l_i[r] = l_i[r] * al + ps;
      m_i[r] = mnew;
      acc0[r] *= al; acc1[r] *= al;
      int prow = r + hiOff;
      pbuf[h][prow * 32 + colL]      = f2bf(p0);
      pbuf[h][prow * 32 + 16 + colL] = f2bf(p1);
    }
    __syncthreads();   // C-layout -> A-layout round-trip through LDS

    v16bf pa, v0, v1;
    {
      int prow = lane & 15;
      for (int j = 0; j < 16; ++j) {
        int k = kOf(j, lane);                         // k == m offset 0..31
        pa[j] = pbuf[h][prow * 32 + k];
        v0[j] = sbf[(mb + k) * DD + h * HID + colL];
        v1[j] = sbf[(mb + k) * DD + h * HID + 16 + colL];
      }
    }
    acc0 = __builtin_amdgcn_wmma_f32_16x16x32_bf16(false, pa, false, v0, (short)0, acc0, false, false);
    acc1 = __builtin_amdgcn_wmma_f32_16x16x32_bf16(false, pa, false, v1, (short)0, acc1, false, false);
    __syncthreads();
  }

  // write normalized aggregation to LDS (self-loop guarantees l_i > 0)
  for (int r = 0; r < 8; ++r) {
    int prow = r + hiOff;
    float inv = 1.f / l_i[r];
    obuf[prow * DD + h * HID + colL]      = acc0[r] * inv;
    obuf[prow * DD + h * HID + 16 + colL] = acc1[r] * inv;
  }
  __syncthreads();

  // lrelu + LayerNorm across all heads + residual: 8 threads/row, 16 cols each
  {
    int row = t >> 3;
    int c0  = (t & 7) * 16;
    float x[16];
    float s = 0.f;
    for (int i = 0; i < 16; ++i) {
      int c = c0 + i;
      x[i] = lrelu(recv[(n0 + row) * DD + c] + obuf[row * DD + c]);
      s += x[i];
    }
    for (int m = 1; m < 8; m <<= 1) s += __shfl_xor(s, m, 8);
    float mean = s / DD;
    float v = 0.f;
    for (int i = 0; i < 16; ++i) { float d = x[i] - mean; v += d * d; }
    for (int m = 1; m < 8; m <<= 1) v += __shfl_xor(v, m, 8);
    float rs = rsqrtf(v / DD + EPSF);
    for (int i = 0; i < 16; ++i) {
      int c = c0 + i;
      out_node[(n0 + row) * DD + c] =
          (x[i] - mean) * rs * g_node[c] + b_node[c] + node[(n0 + row) * DD + c];
    }
  }
}

extern "C" void kernel_launch(void* const* d_in, const int* in_sizes, int n_in,
                              void* d_out, int out_size, void* d_ws, size_t ws_size,
                              hipStream_t stream) {
  const float* node = (const float*)d_in[0];
  const float* recv = (const float*)d_in[1];
  const float* send = (const float*)d_in[2];
  const float* adj  = (const float*)d_in[3];
  const float* Wq = (const float*)d_in[4];
  const float* Wr = (const float*)d_in[5];
  const float* br = (const float*)d_in[6];
  const float* Ws = (const float*)d_in[7];
  const float* bs = (const float*)d_in[8];
  const float* Wm = (const float*)d_in[9];
  const float* bm = (const float*)d_in[10];
  const float* g_node = (const float*)d_in[11];
  const float* b_node = (const float*)d_in[12];
  const float* g_send = (const float*)d_in[13];
  const float* b_send = (const float*)d_in[14];
  const float* g_recv = (const float*)d_in[15];
  const float* b_recv = (const float*)d_in[16];

  float* out_node = (float*)d_out;
  float* out_recv = out_node + (size_t)NN * DD;
  float* out_send = out_recv + (size_t)NN * DD;

  char* ws = (char*)d_ws;
  __bf16* qbf = (__bf16*)ws;                          // 1 MB
  __bf16* sbf = (__bf16*)(ws + (size_t)NN * DD * 2);  // 1 MB
  __bf16* stb = (__bf16*)(ws + (size_t)NN * DD * 4);  // 1 MB
  float*  qrs = (float*) (ws + (size_t)NN * DD * 6);  // 64 KB

  proj_kernel<<<NN / 16, 256, 0, stream>>>(node, recv, send, Wq, Wr, br, Ws, bs, Wm, bm,
                                           g_send, b_send, g_recv, b_recv,
                                           out_recv, out_send, qbf, sbf, stb, qrs);
  attn_kernel<<<NN / 16, 128, 0, stream>>>(node, recv, adj, qbf, sbf, stb, qrs,
                                           g_node, b_node, out_node);
}
